// LocalCorrelation_5738076307585
// MI455X (gfx1250) — matrix operand order repaired and checked by reading, not compiled, not run on hardware
//
#include <hip/hip_runtime.h>
#include <hip/hip_bf16.h>
#include <stdint.h>

typedef __attribute__((ext_vector_type(16))) __bf16 v16bf;
typedef __attribute__((ext_vector_type(8)))  float  v8f;
typedef __attribute__((ext_vector_type(4)))  float  v4f;
typedef __attribute__((ext_vector_type(4)))  __bf16 v4bf;

#define RADIUS   6
#define KTAPS    13
#define C_DIM    256
#define H_DIM    128
#define W_DIM    128
#define HW       (H_DIM * W_DIM)

#define PITCH    264                     // bf16 elems per LDS row (528 B, 16B-aligned)
#define ROWB     (PITCH * 2)             // 528 bytes
#define A_BASE   0
#define A_SIZE   (128 * ROWB)            // 67584 per image row
#define B_BASE   (2 * A_SIZE)            // 135168
#define B_ROWS   144                     // u in [-6,137] -> iu = u+6 in [0,143]
#define B_SIZE   (B_ROWS * ROWB)         // 76032
#define SCR_BASE (B_BASE + B_SIZE)       // 211200
#define SMEM_TOTAL (SCR_BASE + 8 * 2048) // 227584 bytes (<320KB WGP LDS)

// Transpose-convert one image row (256 ch x 128 px) from global f32 [c][w]
// into LDS bf16 [w][c].  Coalesced global reads (lanes = consecutive w);
// packed bf16 conversion (v_cvt_pk_bf16_f32 path); 8-byte LDS writes.
static __device__ __forceinline__ void fill_row_bf16(
    const float* __restrict__ src,    // element (c,w) at src[c*HW + w]
    char* lds_base, int row_off, int tid)
{
  const int wl = tid & 31;
  const int cq = tid >> 5;                        // 0..7
  #pragma unroll
  for (int cc = 0; cc < 8; ++cc) {
    const int cb = (cq + 8 * cc) * 4;             // 0,4,...,252
    const float* s0 = src + (size_t)cb * HW;
    #pragma unroll
    for (int wb = 0; wb < 4; ++wb) {
      const int w = wl + 32 * wb;
      v4f f;
      f.x = s0[w];
      f.y = s0[w + HW];
      f.z = s0[w + 2 * HW];
      f.w = s0[w + 3 * HW];
      v4bf b4 = __builtin_convertvector(f, v4bf);  // RNE, packed cvt
      *(uint2*)(lds_base + (size_t)(w + row_off) * ROWB + cb * 2) =
          __builtin_bit_cast(uint2, b4);
    }
  }
}

// Banded-GEMM compute for one (h, di): 16 WMMAs, band gather via per-wave
// LDS scratch, coalesced stores of 16 px x 13 taps.
static __device__ __forceinline__ void compute_and_store(
    const v16bf* __restrict__ afrag, const char* __restrict__ bbase,
    float* __restrict__ scrW, float* __restrict__ outBase,
    int g, int m, int lh)
{
  v8f acc0 = {}; v8f acc1 = {};
  const char* bb0 = bbase + (size_t)(16 * g + m) * ROWB + 32 * lh;
  const char* bb1 = bb0 + 16 * ROWB;
  #pragma unroll
  for (int kc = 0; kc < 8; ++kc) {
    union { uint4 q[2]; v16bf v; } u0, u1;
    u0.q[0] = *(const uint4*)(bb0 + 64 * kc);
    u0.q[1] = *(const uint4*)(bb0 + 64 * kc + 16);
    u1.q[0] = *(const uint4*)(bb1 + 64 * kc);
    u1.q[1] = *(const uint4*)(bb1 + 64 * kc + 16);
    acc0 = __builtin_amdgcn_wmma_f32_16x16x32_bf16(
        false, afrag[kc], false, u0.v, (short)0, acc0, false, false);
    acc1 = __builtin_amdgcn_wmma_f32_16x16x32_bf16(
        false, afrag[kc], false, u1.v, (short)0, acc1, false, false);
  }
  // D layout -> scratch [tile][M][N]
  #pragma unroll
  for (int r = 0; r < 8; ++r) {
    scrW[(8 * lh + r) * 16 + m]       = acc0[r];
    scrW[256 + (8 * lh + r) * 16 + m] = acc1[r];
  }
  // band gather S[m,j] = D[m, m+j]
  #pragma unroll
  for (int jj = 0; jj < KTAPS; jj += 2) {
    int j = jj + lh;
    if (j < KTAPS) {
      int q = m + j;                               // 0..27
      outBase[(size_t)j * HW + 16 * g + m] =
          scrW[(q >> 4) * 256 + m * 16 + (q & 15)] * 0.0625f;  // 1/sqrt(256)
    }
  }
}

static __device__ __forceinline__ void store_zeros(
    float* __restrict__ outBase, int g, int m, int lh)
{
  #pragma unroll
  for (int jj = 0; jj < KTAPS; jj += 2) {
    int j = jj + lh;
    if (j < KTAPS) outBase[(size_t)j * HW + 16 * g + m] = 0.0f;
  }
}

__global__ __launch_bounds__(256) void localcorr_wmma_kernel(
    const float* __restrict__ zt, const float* __restrict__ zt1,
    float* __restrict__ out)
{
  extern __shared__ char smem[];
  const int tid  = threadIdx.x;
  const int h0   = blockIdx.x * 2;       // h-pair {h0, h0+1}
  const int h1   = h0 + 1;
  const int b    = blockIdx.y;           // 0..7
  const int wave = tid >> 5;             // 0..7 == pixel group g
  const int lane = tid & 31;
  const int m    = lane & 15;
  const int lh   = lane >> 4;
  const int g    = wave;

  // ---- stage both z_t rows (A matrices), zero B border rows ----
  const float* ztBase = zt + (size_t)b * C_DIM * HW;
  fill_row_bf16(ztBase + (size_t)h0 * W_DIM, smem + A_BASE, 0, tid);
  fill_row_bf16(ztBase + (size_t)h1 * W_DIM, smem + A_BASE + A_SIZE, 0, tid);

  for (int idx = tid; idx < 16 * (ROWB / 4); idx += 256) {
    int r  = idx / (ROWB / 4);
    int c4 = idx % (ROWB / 4);
    int iu = (r < 6) ? r : (128 + r);    // rows 0..5 and 134..143 stay zero
    *(uint32_t*)(smem + B_BASE + (size_t)iu * ROWB + c4 * 4) = 0;
  }
  __syncthreads();

  // ---- preload A fragments for both rows (2 x 8 x 8 VGPRs) ----
  v16bf afrag[2][8];
  #pragma unroll
  for (int hs = 0; hs < 2; ++hs) {
    const char* ab = smem + A_BASE + hs * A_SIZE
                   + (size_t)(16 * g + m) * ROWB + lh * 32;
    #pragma unroll
    for (int kc = 0; kc < 8; ++kc) {
      union { uint4 q[2]; v16bf v; } u;
      u.q[0] = *(const uint4*)(ab + 64 * kc);
      u.q[1] = *(const uint4*)(ab + 64 * kc + 16);
      afrag[hs][kc] = u.v;
    }
  }

  float* scrW = (float*)(smem + SCR_BASE + wave * 2048);
  const float* zt1Base = zt1 + (size_t)b * C_DIM * HW;
  char* Bbuf = smem + B_BASE;

  // neighbor rows r = h0-6 .. h0+7 each serve (h0, di=dr) and (h1, di=dr-1)
  for (int dr = 0; dr < KTAPS + 1; ++dr) {
    const int r = h0 + dr - RADIUS;
    const bool rvalid = (r >= 0) && (r < H_DIM);    // block-uniform
    if (rvalid) {
      __syncthreads();                              // prev B consumed
      fill_row_bf16(zt1Base + (size_t)r * W_DIM, Bbuf, RADIUS, tid);
      __syncthreads();                              // B visible
    }
    if (dr <= KTAPS - 1) {                          // (h0, di = dr)
      float* ob = out + (((size_t)b * (KTAPS * KTAPS) + KTAPS * dr) * H_DIM + h0) * W_DIM;
      if (rvalid) compute_and_store(afrag[0], Bbuf, scrW, ob, g, m, lh);
      else        store_zeros(ob, g, m, lh);
    }
    if (dr >= 1) {                                  // (h1, di = dr-1)
      float* ob = out + (((size_t)b * (KTAPS * KTAPS) + KTAPS * (dr - 1)) * H_DIM + h1) * W_DIM;
      if (rvalid) compute_and_store(afrag[1], Bbuf, scrW, ob, g, m, lh);
      else        store_zeros(ob, g, m, lh);
    }
  }
}

extern "C" void kernel_launch(void* const* d_in, const int* in_sizes, int n_in,
                              void* d_out, int out_size, void* d_ws, size_t ws_size,
                              hipStream_t stream) {
  (void)in_sizes; (void)n_in; (void)d_ws; (void)ws_size; (void)out_size;
  const float* zt  = (const float*)d_in[0];
  const float* zt1 = (const float*)d_in[1];
  float* out = (float*)d_out;

  (void)hipFuncSetAttribute((const void*)localcorr_wmma_kernel,
                            hipFuncAttributeMaxDynamicSharedMemorySize,
                            SMEM_TOTAL);
  dim3 grid(H_DIM / 2, 8);   // (h-pair, b)
  dim3 block(256);           // 8 wave32 = 8 pixel groups of 16
  localcorr_wmma_kernel<<<grid, block, SMEM_TOTAL, stream>>>(zt, zt1, out);
}
